// MyModel_87522843560708
// MI455X (gfx1250) — matrix operand order repaired and checked by reading, not compiled, not run on hardware
//
#include <hip/hip_runtime.h>

// ---------------------------------------------------------------------------
// LSTM (B=256, T=1024, F=64, H=32) + TimeDistributed Dense(V=13) + softmax
// MI455X / gfx1250: wave32, WMMA f32_16x16x32_f16 (f16 in, f32 accumulate)
//
// z_t = [x_t | h_{t-1}] @ [W; U] + b   computed as K=96 chain of 3 WMMAs.
// 16 workgroups x 16 batch rows; 8 waves/WG, one 16-col N-tile each.
// Split s_barrier_signal/s_barrier_wait with x staging + next global load
// issued inside the signal->wait window; hardware v_tanh_f32 gate math.
// ---------------------------------------------------------------------------

typedef _Float16 v16h __attribute__((ext_vector_type(16)));
typedef _Float16 v8h  __attribute__((ext_vector_type(8)));
typedef float    v8f  __attribute__((ext_vector_type(8)));

#define B_   256
#define T_   1024
#define F_   64
#define H_   32
#define V_   13
#define K4H  128   // 4*H

#if defined(__has_builtin)
#if __has_builtin(__builtin_amdgcn_tanhf)
#define HAS_HW_TANH 1
#endif
#endif

__device__ __forceinline__ float rcp_f(float x) {
    return __builtin_amdgcn_rcpf(x);            // v_rcp_f32
}

#ifdef HAS_HW_TANH
__device__ __forceinline__ float tanh_f(float x) {
    return __builtin_amdgcn_tanhf(x);           // v_tanh_f32
}
__device__ __forceinline__ float sigmoid_f(float x) {
    // sigmoid(x) = 0.5*tanh(x/2) + 0.5 : one trans op + fma
    return fmaf(0.5f, __builtin_amdgcn_tanhf(0.5f * x), 0.5f);
}
#else
__device__ __forceinline__ float tanh_f(float x) {
    return fmaf(-2.f, rcp_f(__expf(2.f * x) + 1.f), 1.f);
}
__device__ __forceinline__ float sigmoid_f(float x) {
    return rcp_f(1.f + __expf(-x));
}
#endif

// Split work-group barrier pieces (CDNA5 S_BARRIER_SIGNAL/WAIT, barrier -1).
// s_wait_dscnt 0 before signal makes this wave's LDS stores visible to the
// waves that pass the matching wait.
__device__ __forceinline__ void wg_barrier_signal_lds() {
    asm volatile("s_wait_dscnt 0x0\n\ts_barrier_signal -1" ::: "memory");
}
__device__ __forceinline__ void wg_barrier_wait() {
    asm volatile("s_barrier_wait -1" ::: "memory");
}

// Gather a 16x32 f16 A-tile from a row-major f16 buffer (pitch in halves)
// into the WMMA A layout:
//   lane<16 : M=lane,    halves 0..7 = K kcol0+0..7,  halves 8..15 = K kcol0+16..23
//   lane>=16: M=lane-16, halves 0..7 = K kcol0+8..15, halves 8..15 = K kcol0+24..31
__device__ __forceinline__ v16h load_a_tile(const _Float16* src, int pitch,
                                            int kcol0, int lane) {
    const int m  = lane & 15;
    const int kb = (lane >> 4) * 8;
    v8h lo = *(const v8h*)(src + m * pitch + kcol0 + kb);
    v8h hi = *(const v8h*)(src + m * pitch + kcol0 + 16 + kb);
    v16h a;
#pragma unroll
    for (int i = 0; i < 8; ++i) { a[i] = lo[i]; a[i + 8] = hi[i]; }
    return a;
}

// Gather a 32x16 f16 B-tile from a row-major [K][128] f16 buffer into the
// WMMA B layout: col=lane&15, K=koff+h (h<8), koff+16+(h-8) (h>=8); koff=(lane>>4)*8.
__device__ __forceinline__ v16h load_b_tile(const _Float16* src, int krow0,
                                            int ncol0, int lane) {
    const int col  = ncol0 + (lane & 15);
    const int koff = (lane >> 4) * 8;
    v16h bt;
#pragma unroll
    for (int h = 0; h < 16; ++h)
        bt[h] = src[(krow0 + koff + h + (h >= 8 ? 8 : 0)) * K4H + col];
    return bt;
}

__global__ __launch_bounds__(256, 1)
void lstm_fused_kernel(const float* __restrict__ x,   // [B,T,F]
                       const float* __restrict__ W,   // [F,4H]
                       const float* __restrict__ U,   // [H,4H]
                       const float* __restrict__ b,   // [4H]
                       _Float16* __restrict__ hs)     // [B,T,H] f16 out
{
    __shared__ __align__(16) _Float16 lds_wu[(F_ + H_) * K4H]; // [96][128] f16
    __shared__ __align__(16) _Float16 lds_x0[16 * F_];         // x tile ping
    __shared__ __align__(16) _Float16 lds_x1[16 * F_];         // x tile pong
    __shared__ __align__(16) _Float16 lds_h[16 * H_];          // h state f16
    __shared__ __align__(16) float    lds_z[16 * K4H];         // gate preacts

    const int tid  = threadIdx.x;
    const int lane = tid & 31;
    const int wid  = tid >> 5;          // 0..7 -> N-tile
    const int b0   = blockIdx.x * 16;   // batch-row base

    // Stage W (rows 0..63) and U (rows 64..95) as f16 row-major [96][128]
    for (int i = tid; i < F_ * K4H; i += 256) lds_wu[i] = (_Float16)W[i];
    for (int i = tid; i < H_ * K4H; i += 256) lds_wu[F_ * K4H + i] = (_Float16)U[i];
    for (int i = tid; i < 16 * H_; i += 256) lds_h[i] = (_Float16)0.f;

    // x-tile loader mapping: each thread owns 4 consecutive floats of the tile
    const int xr = tid >> 4;         // 0..15
    const int xc = (tid & 15) * 4;   // 0..60
    const size_t xrow = (size_t)(b0 + xr) * T_;   // row base in time units

    // Prologue: stage x(0) into ping buffer, issue load of x(1)
    {
        const float4 xv0 = *(const float4*)(x + (xrow + 0) * F_ + xc);
        _Float16* dst = lds_x0 + xr * F_ + xc;
        dst[0] = (_Float16)xv0.x; dst[1] = (_Float16)xv0.y;
        dst[2] = (_Float16)xv0.z; dst[3] = (_Float16)xv0.w;
    }
    float4 xv = *(const float4*)(x + (xrow + 1) * F_ + xc);   // x(t+1) in regs
    __syncthreads();

    // Per-wave constant B-tiles, held in VGPRs for all T steps.
    const int n0 = wid * 16;
    const v16h Bx0 = load_b_tile(lds_wu, 0,  n0, lane);  // W rows 0..31
    const v16h Bx1 = load_b_tile(lds_wu, 32, n0, lane);  // W rows 32..63
    const v16h Bh  = load_b_tile(lds_wu, 64, n0, lane);  // U rows 0..31
    const float bias = b[n0 + (lane & 15)];

    // Per-thread cell state: 2 cells (r0,cc) and (r0+8,cc)
    const int r0 = tid >> 5;        // 0..7
    const int r1 = r0 + 8;          // 8..15
    const int cc = tid & 31;        // h column 0..31
    float cst0 = 0.f, cst1 = 0.f;

    _Float16* xcur = lds_x0;
    _Float16* xnxt = lds_x1;

#pragma unroll 1
    for (int t = 0; t < T_; ++t) {
        // ---- z = bias + [x_t | h] @ [W; U] : K=96 via 3 chained WMMAs -----
        const v16h A0 = load_a_tile(xcur,  F_, 0,  lane);
        const v16h A1 = load_a_tile(xcur,  F_, 32, lane);
        const v16h Ah = load_a_tile(lds_h, H_, 0,  lane);
        v8f acc;
#pragma unroll
        for (int i = 0; i < 8; ++i) acc[i] = bias;
        acc = __builtin_amdgcn_wmma_f32_16x16x32_f16(false, A0, false, Bx0,
                                                     (short)0, acc, false, false);
        acc = __builtin_amdgcn_wmma_f32_16x16x32_f16(false, A1, false, Bx1,
                                                     (short)0, acc, false, false);
        acc = __builtin_amdgcn_wmma_f32_16x16x32_f16(false, Ah, false, Bh,
                                                     (short)0, acc, false, false);

        // ---- spill z tile (C layout -> row-major LDS) ---------------------
        {
            const int zc    = n0 + (lane & 15);
            const int rbase = (lane >> 4) * 8;
#pragma unroll
            for (int v = 0; v < 8; ++v)
                lds_z[(rbase + v) * K4H + zc] = acc[v];
        }

        // ==== barrier 1: split; independent work inside the window =========
        wg_barrier_signal_lds();
        {
            // stage x(t+1) from regs into the other buffer
            _Float16* dst = xnxt + xr * F_ + xc;
            dst[0] = (_Float16)xv.x; dst[1] = (_Float16)xv.y;
            dst[2] = (_Float16)xv.z; dst[3] = (_Float16)xv.w;
            // issue global load for x(t+2); consumed next iteration
            const int tn2 = (t + 2 < T_) ? (t + 2) : (T_ - 1);
            xv = *(const float4*)(x + (xrow + tn2) * F_ + xc);
            if (t + 3 < T_)
                __builtin_prefetch(x + (xrow + t + 3) * F_ + xc, 0, 3);
        }
        wg_barrier_wait();

        // ---- elementwise LSTM cell update (2 cells per thread) ------------
        {
            const float* zr0 = lds_z + r0 * K4H;
            float ig = sigmoid_f(zr0[0 * H_ + cc]);
            float fg = sigmoid_f(zr0[1 * H_ + cc]);
            float gg = tanh_f   (zr0[2 * H_ + cc]);
            float og = sigmoid_f(zr0[3 * H_ + cc]);
            cst0 = fg * cst0 + ig * gg;
            float hv0 = og * tanh_f(cst0);
            lds_h[r0 * H_ + cc] = (_Float16)hv0;
            hs[((size_t)(b0 + r0) * T_ + t) * H_ + cc] = (_Float16)hv0;

            const float* zr1 = lds_z + r1 * K4H;
            float ig1 = sigmoid_f(zr1[0 * H_ + cc]);
            float fg1 = sigmoid_f(zr1[1 * H_ + cc]);
            float gg1 = tanh_f   (zr1[2 * H_ + cc]);
            float og1 = sigmoid_f(zr1[3 * H_ + cc]);
            cst1 = fg1 * cst1 + ig1 * gg1;
            float hv1 = og1 * tanh_f(cst1);
            lds_h[r1 * H_ + cc] = (_Float16)hv1;
            hs[((size_t)(b0 + r1) * T_ + t) * H_ + cc] = (_Float16)hv1;
        }

        // ==== barrier 2: publish h(t) ======================================
        wg_barrier_signal_lds();
        wg_barrier_wait();

        _Float16* tmp = xcur; xcur = xnxt; xnxt = tmp;
    }
}

__global__ __launch_bounds__(256)
void dense_softmax_kernel(const _Float16* __restrict__ hs,  // [B*T, 32] f16
                          const float* __restrict__ Wd,     // [32, 13]
                          const float* __restrict__ bd,     // [13]
                          float* __restrict__ out)          // [B*T, 13]
{
    __shared__ float sWd[H_ * V_];
    __shared__ float sbd[V_];
    const int tid = threadIdx.x;
    for (int i = tid; i < H_ * V_; i += 256) sWd[i] = Wd[i];
    if (tid < V_) sbd[tid] = bd[tid];
    __syncthreads();

    const size_t row = (size_t)blockIdx.x * 256 + tid;   // grid covers B*T

    float hv[H_];
    const v8h* hp = (const v8h*)(hs + row * H_);
#pragma unroll
    for (int q = 0; q < 4; ++q) {
        v8h v = hp[q];
#pragma unroll
        for (int i = 0; i < 8; ++i) hv[q * 8 + i] = (float)v[i];
    }

    float logits[V_];
#pragma unroll
    for (int v = 0; v < V_; ++v) logits[v] = sbd[v];
#pragma unroll
    for (int h = 0; h < H_; ++h) {
        const float hh = hv[h];
#pragma unroll
        for (int v = 0; v < V_; ++v)
            logits[v] = fmaf(hh, sWd[h * V_ + v], logits[v]);
    }

    float m = logits[0];
#pragma unroll
    for (int v = 1; v < V_; ++v) m = fmaxf(m, logits[v]);
    float s = 0.f;
#pragma unroll
    for (int v = 0; v < V_; ++v) { logits[v] = __expf(logits[v] - m); s += logits[v]; }
    const float inv = rcp_f(s);
#pragma unroll
    for (int v = 0; v < V_; ++v) out[row * V_ + v] = logits[v] * inv;
}

extern "C" void kernel_launch(void* const* d_in, const int* in_sizes, int n_in,
                              void* d_out, int out_size, void* d_ws, size_t ws_size,
                              hipStream_t stream) {
    const float* x  = (const float*)d_in[0];   // [B,T,F]
    const float* W  = (const float*)d_in[1];   // [F,4H]
    const float* U  = (const float*)d_in[2];   // [H,4H]
    const float* b  = (const float*)d_in[3];   // [4H]
    const float* Wd = (const float*)d_in[4];   // [H,V]
    const float* bd = (const float*)d_in[5];   // [V]
    float* out = (float*)d_out;                // [B,T,V]

    _Float16* hs = (_Float16*)d_ws;            // [B,T,H] f16, 16 MB scratch

    lstm_fused_kernel<<<B_ / 16, 256, 0, stream>>>(x, W, U, b, hs);
    dense_softmax_kernel<<<(B_ * T_) / 256, 256, 0, stream>>>(hs, Wd, bd, out);
}
